// Net_64965675319484
// MI455X (gfx1250) — compile-verified, matrix operand order checked by reading, not run on hardware
//
#include <hip/hip_runtime.h>
#include <hip/hip_bf16.h>
#include <math.h>

// ---------------- CDNA5 WMMA types ----------------
typedef __attribute__((ext_vector_type(16))) __bf16 v16bf;
typedef __attribute__((ext_vector_type(8)))  float  v8f;

__device__ __forceinline__ v8f wmma_bf16(v16bf a, v16bf b, v8f c) {
  // D = A(16x32 bf16) x B(32x16 bf16) + C(16x16 f32)
  return __builtin_amdgcn_wmma_f32_16x16x32_bf16(false, a, false, b, (short)0, c,
                                                 false, false);
}

// ---------------- problem constants ----------------
#define BSZ      16
#define NEXO     3
#define CCH      384
#define HGT      28
#define WID      28
#define HWS      784
#define NIMG     48          // BSZ*NEXO
#define NTOK     37632       // NIMG*HWS
#define HID      1536
#define NAFF     6
#define KCLU     3
#define PPTS     2352        // NEXO*HWS
#define KM_IT    50

// ---------------- block reductions (blockDim.x == 256) ----------------
__device__ __forceinline__ float bred_sum(float v, float* red) {
  int t = threadIdx.x; red[t] = v; __syncthreads();
  for (int o = 128; o > 0; o >>= 1) { if (t < o) red[t] += red[t + o]; __syncthreads(); }
  float r = red[0]; __syncthreads(); return r;
}
__device__ __forceinline__ float bred_min(float v, float* red) {
  int t = threadIdx.x; red[t] = v; __syncthreads();
  for (int o = 128; o > 0; o >>= 1) { if (t < o) red[t] = fminf(red[t], red[t + o]); __syncthreads(); }
  float r = red[0]; __syncthreads(); return r;
}
__device__ __forceinline__ float bred_max(float v, float* red) {
  int t = threadIdx.x; red[t] = v; __syncthreads();
  for (int o = 128; o > 0; o >>= 1) { if (t < o) red[t] = fmaxf(red[t], red[t + o]); __syncthreads(); }
  float r = red[0]; __syncthreads(); return r;
}

// ---------------- K0: weight transpose + bf16 convert ----------------
// fc1T[n][k] (1536x384), fc2T[n][k] (384x1536), cXT[tap][o][i] (9x384x384)
__global__ void prep_kernel(const float* __restrict__ fc1_w, const float* __restrict__ fc2_w,
                            const float* __restrict__ c1_w,  const float* __restrict__ c2_w,
                            __bf16* fc1T, __bf16* fc2T, __bf16* c1T, __bf16* c2T) {
  int i = blockIdx.x * 256 + threadIdx.x;
  const int N1 = HID * CCH;          // 589824
  const int N3 = 9 * CCH * CCH;      // 1327104
  if (i < N1) { int n = i / CCH, k = i % CCH; fc1T[i] = (__bf16)fc1_w[k * HID + n]; return; }
  i -= N1;
  if (i < N1) { int n = i / HID, k = i % HID; fc2T[i] = (__bf16)fc2_w[k * CCH + n]; return; }
  i -= N1;
  if (i < N3) { int t = i / (CCH * CCH); int r = i % (CCH * CCH); int o = r / CCH, ic = r % CCH;
                c1T[i] = (__bf16)c1_w[(o * CCH + ic) * 9 + t]; return; }
  i -= N3;
  if (i < N3) { int t = i / (CCH * CCH); int r = i % (CCH * CCH); int o = r / CCH, ic = r % CCH;
                c2T[i] = (__bf16)c2_w[(o * CCH + ic) * 9 + t]; return; }
}

// ---------------- K1: LayerNorm -> bf16 tokens (1 block / token, 128 thr) ----------------
__global__ void ln_kernel(const float* __restrict__ x, const float* __restrict__ g,
                          const float* __restrict__ b, __bf16* __restrict__ xn) {
  __shared__ float red[128];
  const int t = blockIdx.x, tid = threadIdx.x;
  const float* row = x + (size_t)t * CCH;
  float v0 = row[tid], v1 = row[tid + 128], v2 = row[tid + 256];
  red[tid] = v0 + v1 + v2; __syncthreads();
  for (int o = 64; o > 0; o >>= 1) { if (tid < o) red[tid] += red[tid + o]; __syncthreads(); }
  float mu = red[0] * (1.f / CCH); __syncthreads();
  float d0 = v0 - mu, d1 = v1 - mu, d2 = v2 - mu;
  red[tid] = d0 * d0 + d1 * d1 + d2 * d2; __syncthreads();
  for (int o = 64; o > 0; o >>= 1) { if (tid < o) red[tid] += red[tid + o]; __syncthreads(); }
  float inv = rsqrtf(red[0] * (1.f / CCH) + 1e-5f);
  __bf16* orow = xn + (size_t)t * CCH;
  orow[tid]       = (__bf16)(d0 * inv * g[tid]       + b[tid]);
  orow[tid + 128] = (__bf16)(d1 * inv * g[tid + 128] + b[tid + 128]);
  orow[tid + 256] = (__bf16)(d2 * inv * g[tid + 256] + b[tid + 256]);
}

// ---------------- WMMA GEMM tile machinery ----------------
// block = 256 thr = 8 wave32, block tile 64(M)x128(N), wave tile 32x32 (2x2 wmma)
#define GEMM_PROLOG(KD)                                                        \
  __shared__ __align__(32) __bf16 As[64 * 32];                                 \
  __shared__ __align__(32) __bf16 Bs[128 * 32];                                \
  const int tid = threadIdx.x, lane = tid & 31, wid = tid >> 5;                \
  const int wm = wid >> 2, wn = wid & 3;                                       \
  const int m0 = blockIdx.y * 64, n0 = blockIdx.x * 128;                       \
  const int half = lane >> 4, l15 = lane & 15;                                 \
  v8f acc[2][2] = {};

#define GEMM_STAGE_B(BT, KD, k0)                                               \
  { int cc = tid >> 1, kg = (tid & 1) << 4;                                    \
    const uint4* s = reinterpret_cast<const uint4*>((BT) + (size_t)(n0 + cc) * (KD) + (k0) + kg); \
    uint4* d = reinterpret_cast<uint4*>(&Bs[cc * 32 + kg]);                    \
    d[0] = s[0]; d[1] = s[1]; }

#define GEMM_MMA()                                                             \
  { v16bf av0 = *reinterpret_cast<const v16bf*>(&As[(wm * 32      + l15) * 32 + half * 16]); \
    v16bf av1 = *reinterpret_cast<const v16bf*>(&As[(wm * 32 + 16 + l15) * 32 + half * 16]); \
    v16bf bv0 = *reinterpret_cast<const v16bf*>(&Bs[(wn * 32      + l15) * 32 + half * 16]); \
    v16bf bv1 = *reinterpret_cast<const v16bf*>(&Bs[(wn * 32 + 16 + l15) * 32 + half * 16]); \
    acc[0][0] = wmma_bf16(av0, bv0, acc[0][0]);                                \
    acc[0][1] = wmma_bf16(av0, bv1, acc[0][1]);                                \
    acc[1][0] = wmma_bf16(av1, bv0, acc[1][0]);                                \
    acc[1][1] = wmma_bf16(av1, bv1, acc[1][1]); }

// ---------------- K2: h = GELU(xn @ fc1_wT + b1), bf16 out ----------------
__global__ void __launch_bounds__(256) gemm_fc1_kernel(const __bf16* __restrict__ A,
    const __bf16* __restrict__ Bt, const float* __restrict__ bias,
    __bf16* __restrict__ out) {
  constexpr int KD = CCH, ND = HID;
  GEMM_PROLOG(KD)
  for (int k0 = 0; k0 < KD; k0 += 32) {
    { int r = tid >> 2, cg = (tid & 3) << 3;
      const __bf16* src = A + (size_t)(m0 + r) * KD + k0 + cg;
      if (k0 + 32 < KD) __builtin_prefetch(src + 32, 0, 1);
      *reinterpret_cast<uint4*>(&As[r * 32 + cg]) = *reinterpret_cast<const uint4*>(src); }
    GEMM_STAGE_B(Bt, KD, k0)
    __syncthreads();
    GEMM_MMA()
    __syncthreads();
  }
  for (int i = 0; i < 2; ++i)
    for (int j = 0; j < 2; ++j) {
      int nn = n0 + wn * 32 + j * 16 + l15;
      float bn = bias[nn];
      for (int e = 0; e < 8; ++e) {
        int mm = m0 + wm * 32 + i * 16 + e + 8 * half;
        float v = acc[i][j][e] + bn;
        float gl = 0.5f * v * (1.f + erff(v * 0.70710678118f));   // exact GELU
        out[(size_t)mm * ND + nn] = (__bf16)gl;
      }
    }
}

// ---------------- K3: x0 = exo_tok + (h @ fc2_wT + b2), bf16 NHWC out ----------------
__global__ void __launch_bounds__(256) gemm_fc2_kernel(const __bf16* __restrict__ A,
    const __bf16* __restrict__ Bt, const float* __restrict__ bias,
    const float* __restrict__ resid, __bf16* __restrict__ out) {
  constexpr int KD = HID, ND = CCH;
  GEMM_PROLOG(KD)
  for (int k0 = 0; k0 < KD; k0 += 32) {
    { int r = tid >> 2, cg = (tid & 3) << 3;
      const __bf16* src = A + (size_t)(m0 + r) * KD + k0 + cg;
      if (k0 + 32 < KD) __builtin_prefetch(src + 32, 0, 1);
      *reinterpret_cast<uint4*>(&As[r * 32 + cg]) = *reinterpret_cast<const uint4*>(src); }
    GEMM_STAGE_B(Bt, KD, k0)
    __syncthreads();
    GEMM_MMA()
    __syncthreads();
  }
  for (int i = 0; i < 2; ++i)
    for (int j = 0; j < 2; ++j) {
      int nn = n0 + wn * 32 + j * 16 + l15;
      float bn = bias[nn];
      for (int e = 0; e < 8; ++e) {
        int mm = m0 + wm * 32 + i * 16 + e + 8 * half;
        float v = acc[i][j][e] + bn + resid[(size_t)mm * ND + nn];
        out[(size_t)mm * ND + nn] = (__bf16)v;
      }
    }
}

// ---------------- K4: 3x3 conv (implicit GEMM, 9 taps) + bias + BN + ReLU ----------------
__global__ void __launch_bounds__(256) conv3x3_kernel(const __bf16* __restrict__ X,
    const __bf16* __restrict__ Wt, const float* __restrict__ cb,
    const float* __restrict__ bg, const float* __restrict__ bb,
    const float* __restrict__ bm, const float* __restrict__ bv,
    __bf16* __restrict__ out) {
  constexpr int KD = CCH, ND = CCH;
  GEMM_PROLOG(KD)
  const int rA = tid >> 2, cgA = (tid & 3) << 3;
  const int p = m0 + rA;
  const int img = p / HWS, hw = p % HWS;
  const int ph = hw / WID, pw = hw % WID;
  for (int t = 0; t < 9; ++t) {
    const int dy = t / 3 - 1, dx = t % 3 - 1;
    const int hh = ph + dy, wp = pw + dx;
    const bool ok = (hh >= 0) && (hh < HGT) && (wp >= 0) && (wp < WID);
    const __bf16* src  = X + ((size_t)img * HWS + hh * WID + wp) * CCH;
    const __bf16* wtap = Wt + (size_t)t * CCH * CCH;
    for (int kc = 0; kc < KD; kc += 32) {
      uint4 val = {0u, 0u, 0u, 0u};
      if (ok) val = *reinterpret_cast<const uint4*>(src + kc + cgA);
      *reinterpret_cast<uint4*>(&As[rA * 32 + cgA]) = val;
      GEMM_STAGE_B(wtap, KD, kc)
      __syncthreads();
      GEMM_MMA()
      __syncthreads();
    }
  }
  for (int i = 0; i < 2; ++i)
    for (int j = 0; j < 2; ++j) {
      int nn = n0 + wn * 32 + j * 16 + l15;
      float bias_n = cb[nn];
      float sc = rsqrtf(bv[nn] + 1e-5f) * bg[nn];
      float sh = bb[nn] - bm[nn] * sc;
      for (int e = 0; e < 8; ++e) {
        int mm = m0 + wm * 32 + i * 16 + e + 8 * half;
        float y = (acc[i][j][e] + bias_n) * sc + sh;
        out[(size_t)mm * ND + nn] = (__bf16)fmaxf(y, 0.f);
      }
    }
}

// ---------------- K5a: 1x1 conv head (384 -> 6), cam[img][a][hw] f32 ----------------
__global__ void affcam_kernel(const __bf16* __restrict__ X, const float* __restrict__ fw,
                              const float* __restrict__ fb, float* __restrict__ cam) {
  __shared__ float wsh[NAFF * CCH];
  const int tid = threadIdx.x;
  for (int i = tid; i < NAFF * CCH; i += 256) wsh[i] = fw[i];
  __syncthreads();
  const int p = blockIdx.x * 256 + tid;           // < NTOK exactly
  const __bf16* row = X + (size_t)p * CCH;
  float a[NAFF];
  for (int q = 0; q < NAFF; ++q) a[q] = fb[q];
  for (int c = 0; c < CCH; ++c) {
    float xv = (float)row[c];
    for (int q = 0; q < NAFF; ++q) a[q] += xv * wsh[q * CCH + c];
  }
  const int img = p / HWS, hw = p % HWS;
  for (int q = 0; q < NAFF; ++q)
    cam[((size_t)img * NAFF + q) * HWS + hw] = a[q];
}

// ---------------- K5b: aff_logits = mean over hw ----------------
__global__ void logits_kernel(const float* __restrict__ cam, float* __restrict__ out) {
  __shared__ float red[256];
  const int tid = threadIdx.x;
  const float* row = cam + (size_t)blockIdx.x * HWS;
  float s = 0.f;
  for (int hw = tid; hw < HWS; hw += 256) s += row[hw];
  float tot = bred_sum(s, red);
  if (tid == 0) out[blockIdx.x] = tot * (1.f / HWS);
}

// ---------------- K5c: gt_aff_cam gather by task label ----------------
__global__ void gt_kernel(const float* __restrict__ cam, const int* __restrict__ task,
                          float* __restrict__ gt) {
  const int idx = blockIdx.x * 256 + threadIdx.x;   // < NTOK (== BSZ*NEXO*HWS)
  const int b = idx / PPTS, r = idx % PPTS;
  const int e = r / HWS, hw = r % HWS;
  const int lbl = task[b];
  gt[idx] = cam[(((size_t)(b * NEXO + e)) * NAFF + lbl) * HWS + hw];
}

// ---------------- K6: ego SAM map ----------------
__global__ void egosam_kernel(const float* __restrict__ attn, float* __restrict__ sam) {
  __shared__ float red[256];
  const int b = blockIdx.x, tid = threadIdx.x;
  const float* a0 = attn + ((size_t)b * 6 + 0) * HWS;
  const float* a1 = attn + ((size_t)b * 6 + 1) * HWS;
  const float* a3 = attn + ((size_t)b * 6 + 3) * HWS;
  float s0 = 0.f, s1 = 0.f, s3 = 0.f;
  for (int hw = tid; hw < HWS; hw += 256) { s0 += a0[hw]; s1 += a1[hw]; s3 += a3[hw]; }
  float m0 = bred_sum(s0, red) * (1.f / HWS);
  float m1 = bred_sum(s1, red) * (1.f / HWS);
  float m3 = bred_sum(s3, red) * (1.f / HWS);
  float lmin = 3.4e38f, lmax = -3.4e38f;
  for (int hw = tid; hw < HWS; hw += 256) {
    float v = (((a0[hw] > m0) ? 1.f : 0.f) + ((a1[hw] > m1) ? 1.f : 0.f) +
               ((a3[hw] > m3) ? 1.f : 0.f)) * (1.f / 3.f);
    lmin = fminf(lmin, v); lmax = fmaxf(lmax, v);
  }
  float lo = bred_min(lmin, red);
  float hi = bred_max(lmax, red);
  float inv = 1.f / (hi - lo + 1e-12f);
  for (int hw = tid; hw < HWS; hw += 256) {
    float v = (((a0[hw] > m0) ? 1.f : 0.f) + ((a1[hw] > m1) ? 1.f : 0.f) +
               ((a3[hw] > m3) ? 1.f : 0.f)) * (1.f / 3.f);
    sam[(size_t)b * HWS + hw] = (v - lo) * inv;
  }
}

// ---------------- K7: per-sample KMeans + similarity + IoU scoring ----------------
__global__ void __launch_bounds__(256) per_sample_kernel(
    const float* __restrict__ ego_desc, const float* __restrict__ exo_desc,
    const float* __restrict__ gt, const float* __restrict__ sam,
    float* __restrict__ out_sim, float* __restrict__ out_exo,
    float* __restrict__ out_p, float* __restrict__ out_proto) {
  const int b = blockIdx.x, tid = threadIdx.x;
  const float* pts  = exo_desc + (size_t)b * PPTS * CCH;   // [P][C]
  const float* egob = ego_desc + (size_t)b * HWS * CCH;    // [HW][C]
  const float* gtb  = gt + (size_t)b * PPTS;               // [3][HW]
  const float* samb = sam + (size_t)b * HWS;

  __shared__ float red[256];
  __shared__ float s_wts[PPTS];
  __shared__ float s_cent[KCLU][CCH];
  __shared__ float s_new[KCLU][CCH];
  __shared__ float s_cnt[KCLU];
  __shared__ unsigned char s_asn[PPTS];
  __shared__ float s_sim[KCLU * HWS];
  __shared__ float s_scal[16];
  __shared__ int   s_sel[KCLU];

  // --- weights from thresholded per-exo min-max of gt cam ---
  for (int e = 0; e < NEXO; ++e) {
    float lmin = 3.4e38f, lmax = -3.4e38f;
    for (int hw = tid; hw < HWS; hw += 256) {
      float v = gtb[e * HWS + hw]; lmin = fminf(lmin, v); lmax = fmaxf(lmax, v);
    }
    float lo = bred_min(lmin, red);
    float hi = bred_max(lmax, red);
    float inv = 1.f / (hi - lo + 1e-10f);
    for (int hw = tid; hw < HWS; hw += 256)
      s_wts[e * HWS + hw] = (((gtb[e * HWS + hw] - lo) * inv) > 0.6f) ? 1.f : 0.f;
  }
  __syncthreads();
  float part = 0.f;
  for (int p = tid; p < PPTS; p += 256) part += s_wts[p];
  float wsum = bred_sum(part, red);
  const bool valid = (wsum >= (float)KCLU);

  // --- deterministic init: first-K by score = wts - idx*1e-6 ---
  if (tid == 0) {
    int k = 0;
    for (int p = 0; p < PPTS && k < KCLU; ++p) if (s_wts[p] > 0.f) s_sel[k++] = p;
    for (int p = 0; p < PPTS && k < KCLU; ++p) if (!(s_wts[p] > 0.f)) s_sel[k++] = p;
  }
  __syncthreads();
  for (int idx = tid; idx < KCLU * CCH; idx += 256) {
    int k = idx / CCH, c = idx % CCH;
    s_cent[k][c] = pts[(size_t)s_sel[k] * CCH + c];
  }
  __syncthreads();

  // --- KMeans iterations ---
  for (int it = 0; it < KM_IT; ++it) {
    for (int idx = tid; idx < KCLU * CCH; idx += 256) s_new[idx / CCH][idx % CCH] = 0.f;
    if (tid < KCLU) {
      s_cnt[tid] = 0.f;
      float cc = 0.f;
      for (int c = 0; c < CCH; ++c) { float v = s_cent[tid][c]; cc += v * v; }
      s_scal[4 + tid] = cc;
    }
    __syncthreads();
    for (int p = tid; p < PPTS; p += 256) {
      const float* pv = pts + (size_t)p * CCH;
      float d0 = 0.f, d1 = 0.f, d2 = 0.f;
      for (int c = 0; c < CCH; ++c) {
        float x = pv[c];
        d0 += x * s_cent[0][c]; d1 += x * s_cent[1][c]; d2 += x * s_cent[2][c];
      }
      float e0 = s_scal[4] - 2.f * d0, e1 = s_scal[5] - 2.f * d1, e2 = s_scal[6] - 2.f * d2;
      int a = 0; float best = e0;
      if (e1 < best) { best = e1; a = 1; }
      if (e2 < best) { best = e2; a = 2; }
      s_asn[p] = (unsigned char)a;
      float w = s_wts[p];
      if (w > 0.f) {
        atomicAdd(&s_cnt[a], w);                        // ds_add_f32
        for (int c = 0; c < CCH; ++c) atomicAdd(&s_new[a][c], pv[c]);
      }
    }
    __syncthreads();
    for (int idx = tid; idx < KCLU * CCH; idx += 256) {
      int k = idx / CCH, c = idx % CCH;
      float cnt = s_cnt[k];
      if (cnt > 0.f) s_cent[k][c] = s_new[k][c] / fmaxf(cnt, 1e-12f);
    }
    __syncthreads();
  }

  // --- l2-normalize centroids ---
  if (tid < KCLU) {
    float nn = 0.f;
    for (int c = 0; c < CCH; ++c) { float v = s_cent[tid][c]; nn += v * v; }
    s_scal[8 + tid] = 1.f / fmaxf(sqrtf(nn), 1e-12f);
  }
  __syncthreads();
  for (int idx = tid; idx < KCLU * CCH; idx += 256)
    s_cent[idx / CCH][idx % CCH] *= s_scal[8 + idx / CCH];
  __syncthreads();

  // --- sim[k][hw] = <clu_k, ego_hw/||ego_hw||> ---
  for (int idx = tid; idx < KCLU * HWS; idx += 256) {
    int k = idx / HWS, hw = idx % HWS;
    const float* ev = egob + (size_t)hw * CCH;
    float dot = 0.f, nn = 0.f;
    for (int c = 0; c < CCH; ++c) { float x = ev[c]; dot += x * s_cent[k][c]; nn += x * x; }
    float sv = dot / fmaxf(sqrtf(nn), 1e-12f);
    s_sim[idx] = sv;
    out_sim[((size_t)b * KCLU + k) * HWS + hw] = valid ? sv : 0.f;
  }
  // --- exo_sim[n][k][hw] ---
  for (int idx = tid; idx < NEXO * KCLU * HWS; idx += 256) {
    int n = idx / (KCLU * HWS), k = (idx / HWS) % KCLU, hw = idx % HWS;
    const float* pv = pts + ((size_t)n * HWS + hw) * CCH;
    float dot = 0.f, nn = 0.f;
    for (int c = 0; c < CCH; ++c) { float x = pv[c]; dot += x * s_cent[k][c]; nn += x * x; }
    float sv = dot / fmaxf(sqrtf(nn), 1e-12f);
    out_exo[(((size_t)b * NEXO + n) * KCLU + k) * HWS + hw] = valid ? sv : 0.f;
  }
  __syncthreads();

  // --- IoU-style part score ---
  float sp = 0.f;
  for (int hw = tid; hw < HWS; hw += 256) sp += samb[hw];
  float sam_mean = bred_sum(sp, red) * (1.f / HWS);
  float scnt = 0.f;
  for (int hw = tid; hw < HWS; hw += 256) scnt += (samb[hw] > sam_mean) ? 1.f : 0.f;
  float ns = bred_sum(scnt, red);

  for (int k = 0; k < KCLU; ++k) {
    float lmin = 3.4e38f, lmax = -3.4e38f, ls = 0.f;
    for (int hw = tid; hw < HWS; hw += 256) {
      float v = s_sim[k * HWS + hw];
      lmin = fminf(lmin, v); lmax = fmaxf(lmax, v); ls += v;
    }
    float lo = bred_min(lmin, red);
    float hi = bred_max(lmax, red);
    float ssum = bred_sum(ls, red);
    float inv = 1.f / (hi - lo + 1e-12f);
    float thr = (ssum * (1.f / HWS) - lo) * inv;
    float li = 0.f, lh = 0.f;
    for (int hw = tid; hw < HWS; hw += 256) {
      float v = (s_sim[k * HWS + hw] - lo) * inv;
      float hd = (v > thr) ? 1.f : 0.f;
      lh += hd;
      li += hd * ((samb[hw] > sam_mean) ? 1.f : 0.f);
    }
    float inter = bred_sum(li, red);
    float nh = bred_sum(lh, red);
    float uni = nh + ns - inter;
    float pk = 0.5f * (inter / (nh + 1e-12f) + ns / (uni + 1e-12f));
    if (tid == 0) s_scal[k] = valid ? pk : 0.f;
    __syncthreads();
  }
  if (tid < KCLU) out_p[(size_t)b * KCLU + tid] = s_scal[tid];
  if (tid == 0) {
    int arg = 0; float pm = s_scal[0];
    if (s_scal[1] > pm) { pm = s_scal[1]; arg = 1; }
    if (s_scal[2] > pm) { pm = s_scal[2]; arg = 2; }
    s_sel[0] = arg;
    s_scal[12] = (valid && pm >= 0.6f) ? 1.f : 0.f;
  }
  __syncthreads();
  const int arg = s_sel[0];
  const float gate = s_scal[12];
  for (int c = tid; c < CCH; c += 256)
    out_proto[(size_t)b * CCH + c] = (gate > 0.f) ? s_cent[arg][c] : 0.f;
}

// ---------------- host launch ----------------
extern "C" void kernel_launch(void* const* d_in, const int* in_sizes, int n_in,
                              void* d_out, int out_size, void* d_ws, size_t ws_size,
                              hipStream_t stream) {
  (void)in_sizes; (void)n_in; (void)out_size; (void)ws_size;
  const float* ego_desc = (const float*)d_in[0];
  const float* exo_desc = (const float*)d_in[1];
  const float* attn     = (const float*)d_in[2];
  const int*   task     = (const int*)  d_in[3];
  const float* ln_g = (const float*)d_in[4];
  const float* ln_b = (const float*)d_in[5];
  const float* fc1_w = (const float*)d_in[6];
  const float* fc1_b = (const float*)d_in[7];
  const float* fc2_w = (const float*)d_in[8];
  const float* fc2_b = (const float*)d_in[9];
  const float* c1_w = (const float*)d_in[10];
  const float* c1_b = (const float*)d_in[11];
  const float* bn1_g = (const float*)d_in[12];
  const float* bn1_b = (const float*)d_in[13];
  const float* bn1_m = (const float*)d_in[14];
  const float* bn1_v = (const float*)d_in[15];
  const float* c2_w = (const float*)d_in[16];
  const float* c2_b = (const float*)d_in[17];
  const float* bn2_g = (const float*)d_in[18];
  const float* bn2_b = (const float*)d_in[19];
  const float* bn2_m = (const float*)d_in[20];
  const float* bn2_v = (const float*)d_in[21];
  const float* fc_w = (const float*)d_in[22];
  const float* fc_b = (const float*)d_in[23];
  float* out = (float*)d_out;

  char* ws = (char*)d_ws;
  size_t off = 0;
  auto take = [&](size_t bytes) -> char* {
    char* p = ws + off;
    off += (bytes + 255) & ~(size_t)255;
    return p;
  };
  __bf16* fc1T = (__bf16*)take((size_t)HID * CCH * 2);
  __bf16* fc2T = (__bf16*)take((size_t)HID * CCH * 2);
  __bf16* c1T  = (__bf16*)take((size_t)9 * CCH * CCH * 2);
  __bf16* c2T  = (__bf16*)take((size_t)9 * CCH * CCH * 2);
  __bf16* xn   = (__bf16*)take((size_t)NTOK * CCH * 2);
  __bf16* hbuf = (__bf16*)take((size_t)NTOK * HID * 2);
  __bf16* x0   = (__bf16*)take((size_t)NTOK * CCH * 2);
  __bf16* x1   = (__bf16*)take((size_t)NTOK * CCH * 2);
  __bf16* x2   = (__bf16*)take((size_t)NTOK * CCH * 2);
  float*  cam  = (float*)take((size_t)NIMG * NAFF * HWS * 4);
  float*  gtb  = (float*)take((size_t)BSZ * NEXO * HWS * 4);
  float*  sam  = (float*)take((size_t)BSZ * HWS * 4);

  // weight prep: (2*589824 + 2*1327104) / 256 = 14976 blocks
  prep_kernel<<<14976, 256, 0, stream>>>(fc1_w, fc2_w, c1_w, c2_w, fc1T, fc2T, c1T, c2T);
  ln_kernel<<<NTOK, 128, 0, stream>>>(exo_desc, ln_g, ln_b, xn);
  gemm_fc1_kernel<<<dim3(HID / 128, NTOK / 64), 256, 0, stream>>>(xn, fc1T, fc1_b, hbuf);
  gemm_fc2_kernel<<<dim3(CCH / 128, NTOK / 64), 256, 0, stream>>>(hbuf, fc2T, fc2_b, exo_desc, x0);
  conv3x3_kernel<<<dim3(CCH / 128, NTOK / 64), 256, 0, stream>>>(x0, c1T, c1_b,
      bn1_g, bn1_b, bn1_m, bn1_v, x1);
  conv3x3_kernel<<<dim3(CCH / 128, NTOK / 64), 256, 0, stream>>>(x1, c2T, c2_b,
      bn2_g, bn2_b, bn2_m, bn2_v, x2);
  affcam_kernel<<<NTOK / 256, 256, 0, stream>>>(x2, fc_w, fc_b, cam);
  logits_kernel<<<NIMG * NAFF, 256, 0, stream>>>(cam, out);                 // [0,288)
  gt_kernel<<<NTOK / 256, 256, 0, stream>>>(cam, task, gtb);
  egosam_kernel<<<BSZ, 256, 0, stream>>>(attn, sam);
  per_sample_kernel<<<BSZ, 256, 0, stream>>>(ego_desc, exo_desc, gtb, sam,
      out + 288,                 // sim_maps   [16,3,784]
      out + 288 + 37632,         // exo_sim    [16,3,3,784]
      out + 288 + 37632 + 112896,        // part_score [16,3]
      out + 288 + 37632 + 112896 + 48);  // part_proto [16,384]
}
